// ClaimEvidenceAttentionModel_83150566850803
// MI455X (gfx1250) — matrix operand order repaired and checked by reading, not compiled, not run on hardware
//
#include <hip/hip_runtime.h>
#include <hip/hip_bf16.h>
#include <math.h>

// ---------------------------------------------------------------------------
// ClaimEvidenceAttention fused pipeline for MI455X (gfx1250, wave32, WMMA)
// B=256, K=64, D=1024, H=4, NL=3
// Dominant GEMM (e_proj) + MLP GEMM run on v_wmma_f32_16x16x32_bf16.
// B operand staged TRANSPOSED in LDS so both A and B fragments are 2x
// ds_load_b128 per lane. k3 stages A via global_load_async_to_lds_b128.
// ---------------------------------------------------------------------------

#define B_  256
#define K_  64
#define D_  1024
#define H_  4
#define NL_ 3

typedef __attribute__((ext_vector_type(16))) __bf16 v16bf;
typedef __attribute__((ext_vector_type(8)))  float  v8f;

static __device__ __forceinline__ __bf16 to_bf16(float x) { return (__bf16)x; }

// --- fragment gathers (wave32, contiguous k = halfid*16 + t in A and B) ----------------------------------
// A (16x32 bf16, MxK), lane m = lane&15, halfid = lane>>4:
//   f[0..15] = A[m][halfid*16 + 0..15]          (one contiguous 32B read)
static __device__ __forceinline__ v16bf gather_a(const __bf16* Abuf /*[64][32]*/,
                                                 int m, int halfid) {
  v16bf f;
  const __bf16* p = Abuf + m * 32 + halfid * 16;
#pragma unroll
  for (int t = 0; t < 16; ++t) f[t] = p[t];
  return f;
}
// B (32x16 bf16, KxN) from TRANSPOSED LDS tile Bsw[n][k] (k contiguous):
//   f[t] = B[k = halfid*16 + t][n]  -> one contiguous 32B run per lane.
static __device__ __forceinline__ v16bf gather_b_sw(const __bf16* Bsw /*[64][32]*/,
                                                    int n, int halfid) {
  v16bf f;
  const __bf16* p = Bsw + n * 32 + halfid * 16;
#pragma unroll
  for (int t = 0; t < 16; ++t) f[t] = p[t];
  return f;
}

static __device__ __forceinline__ v8f wmma_bf16(v16bf a, v16bf b, v8f c) {
  return __builtin_amdgcn_wmma_f32_16x16x32_bf16(false, a, false, b,
                                                 (short)0, c, false, false);
}

// ---------------------------------------------------------------------------
// k0: WcSum[h][d] = sum_e W_c[h][d][e]   ('bd,hde->bhd' does NOT contract d;
//     c_proj[b,h,d] = c_out[b,d] * WcSum[h,d]).  wave-per-row, grid = 512.
// ---------------------------------------------------------------------------
__global__ __launch_bounds__(256) void k0_wcsum(const float* __restrict__ Wc,
                                                float* __restrict__ wcsum) {
  int row  = blockIdx.x * 8 + (threadIdx.x >> 5);
  int lane = threadIdx.x & 31;
  const float* p = Wc + (size_t)row * D_;
  float s = 0.f;
  for (int e = lane; e < D_; e += 32) s += p[e];
#pragma unroll
  for (int off = 16; off > 0; off >>= 1) s += __shfl_xor(s, off, 32);
  if (lane == 0) wcsum[row] = s;
}

// ---------------------------------------------------------------------------
// k1: per (b,h): fused e_proj GEMM (64x1024x1024 bf16 WMMA) -> tanh*v -> u ->
//     softmax (alpha -> d_out) -> e_att_heads (ws) -> gate evidence dot (ws)
// grid = B*H = 1024 blocks, 256 threads; waves laid out 4(M) x 2(N-half),
// each wave owns TWO 16x16 C tiles (N-chunk = 64, A fragment reused 2x).
// ---------------------------------------------------------------------------
__global__ __launch_bounds__(256)
void k1_attention(const float* __restrict__ c_out,
                  const float* __restrict__ e_emb,
                  const int*   __restrict__ emask,
                  const float* __restrict__ W_e,
                  const float* __restrict__ v,
                  const float* __restrict__ gate_w,
                  const float* __restrict__ wcsum,
                  float* __restrict__ eatt_ws,   // [B][H][D]
                  float* __restrict__ edot_ws,   // [B][H]
                  float* __restrict__ alpha_out) // d_out + B*NL
{
  __shared__ __bf16 Abuf[64 * 32];   // [m][k]
  __shared__ __bf16 Bsw[64 * 32];    // [n][k] (transposed tile)
  __shared__ float  u_lds[K_];
  __shared__ float  alpha_lds[K_];
  __shared__ float  red[256];

  const int tid    = threadIdx.x;
  const int b      = blockIdx.x >> 2;
  const int h      = blockIdx.x & 3;
  const int wave   = tid >> 5;
  const int lane   = tid & 31;
  const int mtile  = wave & 3;    // 0..3
  const int nhalf  = wave >> 2;   // 0..1
  const int lane15 = lane & 15;
  const int halfid = lane >> 4;

  if (tid < K_) u_lds[tid] = 0.f;

  const float* Arow = e_emb + (size_t)b * K_ * D_;
  const float* Brow = W_e + (size_t)h * D_ * D_;

  for (int nc = 0; nc < D_ / 64; ++nc) {
    v8f acc0 = {0.f, 0.f, 0.f, 0.f, 0.f, 0.f, 0.f, 0.f};
    v8f acc1 = {0.f, 0.f, 0.f, 0.f, 0.f, 0.f, 0.f, 0.f};
    for (int kd = 0; kd < D_ / 32; ++kd) {
      __syncthreads();
      // stage A chunk (64m x 32k of e_emb[b]) -> bf16 LDS, coalesced
      {
        int m   = tid >> 2;
        int kk0 = (tid & 3) * 8;
        const float* src = Arow + (size_t)m * D_ + kd * 32 + kk0;
#pragma unroll
        for (int i = 0; i < 8; ++i) Abuf[m * 32 + kk0 + i] = to_bf16(src[i]);
      }
      // stage B chunk (32k x 64n of W_e[h]) -> TRANSPOSED bf16 LDS Bsw[n][k]
      {
        int kk  = tid >> 3;
        int nn0 = (tid & 7) * 8;
        const float* src = Brow + (size_t)(kd * 32 + kk) * D_ + nc * 64 + nn0;
#pragma unroll
        for (int i = 0; i < 8; ++i) Bsw[(nn0 + i) * 32 + kk] = to_bf16(src[i]);
        if (kd + 1 < D_ / 32)
          __builtin_prefetch(src + 32 * D_, 0, 1);  // next k-slab of W_e
      }
      __syncthreads();
      v16bf af  = gather_a(Abuf, mtile * 16 + lane15, halfid);
      v16bf bf0 = gather_b_sw(Bsw, nhalf * 32 + lane15, halfid);
      v16bf bf1 = gather_b_sw(Bsw, nhalf * 32 + 16 + lane15, halfid);
      acc0 = wmma_bf16(af, bf0, acc0);
      acc1 = wmma_bf16(af, bf1, acc1);
    }
    // epilogue: u[k] += sum_e tanh(c_proj[e] + e_proj[k][e]) * v[h][e]
    int   e0  = nc * 64 + nhalf * 32 + lane15;
    int   e1  = e0 + 16;
    float cw0 = c_out[(size_t)b * D_ + e0] * wcsum[(size_t)h * D_ + e0];
    float cw1 = c_out[(size_t)b * D_ + e1] * wcsum[(size_t)h * D_ + e1];
    float vv0 = v[(size_t)h * D_ + e0];
    float vv1 = v[(size_t)h * D_ + e1];
#pragma unroll
    for (int j = 0; j < 8; ++j) {
      int   m   = mtile * 16 + halfid * 8 + j;
      float val = tanhf(cw0 + acc0[j]) * vv0 + tanhf(cw1 + acc1[j]) * vv1;
#pragma unroll
      for (int off = 8; off > 0; off >>= 1) val += __shfl_xor(val, off, 16);
      if (lane15 == 0) atomicAdd(&u_lds[m], val);
    }
  }
  __syncthreads();

  // softmax over k (K=64; serialized on thread 0, negligible)
  if (tid == 0) {
    const int* mk = emask + (size_t)b * K_;
    float mx = -__builtin_inff();
    for (int k = 0; k < K_; ++k) {
      float uk = (mk[k] == 0) ? -__builtin_inff() : u_lds[k];
      u_lds[k] = uk;
      mx = fmaxf(mx, uk);
    }
    float ssum = 0.f;
    for (int k = 0; k < K_; ++k) {
      float ex = (mk[k] == 0) ? 0.f : __expf(u_lds[k] - mx);
      alpha_lds[k] = ex;
      ssum += ex;
    }
    float inv = 1.f / ssum;
    for (int k = 0; k < K_; ++k) alpha_lds[k] *= inv;
  }
  __syncthreads();
  if (tid < K_) alpha_out[((size_t)(b * H_ + h)) * K_ + tid] = alpha_lds[tid];

  // e_att_heads[b,h,:] = alpha @ e_emb[b]  (+ gate evidence dot)
  float gpart = 0.f;
  const float* gwE = gate_w + D_;
#pragma unroll
  for (int c = 0; c < 4; ++c) {
    int   d    = tid + c * 256;
    float accd = 0.f;
    for (int k = 0; k < K_; ++k) accd += alpha_lds[k] * Arow[(size_t)k * D_ + d];
    eatt_ws[((size_t)b * H_ + h) * D_ + d] = accd;
    gpart += accd * gwE[d];
  }
  red[tid] = gpart;
  __syncthreads();
  for (int s = 128; s > 0; s >>= 1) {
    if (tid < s) red[tid] += red[tid + s];
    __syncthreads();
  }
  if (tid == 0) edot_ws[b * H_ + h] = red[0];
}

// ---------------------------------------------------------------------------
// k2: per b: gate softmax over heads (head_weights -> d_out), e_att, and
//     features [c, e, |c-e|, c*e] -> bf16 ws.  grid = B, 256 threads
// ---------------------------------------------------------------------------
__global__ __launch_bounds__(256)
void k2_gate(const float* __restrict__ c_out,
             const float* __restrict__ gate_w,
             const float* __restrict__ gate_b,
             const float* __restrict__ eatt_ws,
             const float* __restrict__ edot_ws,
             float* __restrict__ hw_out,    // d_out + B*NL + B*H*K
             __bf16* __restrict__ hfeat)    // [B][4*D] bf16
{
  __shared__ float red[256];
  __shared__ float hw[H_];
  int tid = threadIdx.x;
  int b   = blockIdx.x;
  const float* crow = c_out + (size_t)b * D_;

  float cp = 0.f;
#pragma unroll
  for (int c = 0; c < 4; ++c) {
    int d = tid + c * 256;
    cp += crow[d] * gate_w[d];
  }
  red[tid] = cp;
  __syncthreads();
  for (int s = 128; s > 0; s >>= 1) {
    if (tid < s) red[tid] += red[tid + s];
    __syncthreads();
  }
  if (tid == 0) {
    float cdot = red[0] + gate_b[0];
    float gl[H_];
    float mx = -__builtin_inff();
    for (int hh = 0; hh < H_; ++hh) {
      gl[hh] = cdot + edot_ws[b * H_ + hh];
      mx = fmaxf(mx, gl[hh]);
    }
    float s = 0.f;
    for (int hh = 0; hh < H_; ++hh) { gl[hh] = __expf(gl[hh] - mx); s += gl[hh]; }
    for (int hh = 0; hh < H_; ++hh) {
      hw[hh] = gl[hh] / s;
      hw_out[b * H_ + hh] = hw[hh];
    }
  }
  __syncthreads();

  __bf16* out = hfeat + (size_t)b * 4 * D_;
#pragma unroll
  for (int c = 0; c < 4; ++c) {
    int   d  = tid + c * 256;
    float ea = 0.f;
    for (int hh = 0; hh < H_; ++hh)
      ea += hw[hh] * eatt_ws[((size_t)b * H_ + hh) * D_ + d];
    float cc = crow[d];
    out[d]          = to_bf16(cc);
    out[D_ + d]     = to_bf16(ea);
    out[2 * D_ + d] = to_bf16(fabsf(cc - ea));
    out[3 * D_ + d] = to_bf16(cc * ea);
  }
}

// ---------------------------------------------------------------------------
// k3: hid = relu(hfeat @ w1^T + b1)  -- M=256, N=1024, Kc=4096 bf16 WMMA GEMM
// grid = (256/64) * (1024/64) = 64 blocks, 256 threads; A operand is already
// bf16 in ws, staged with global_load_async_to_lds_b128 (ASYNCcnt path).
// ---------------------------------------------------------------------------
__global__ __launch_bounds__(256)
void k3_mlp1(const __bf16* __restrict__ hfeat,  // [256][4096]
             const float*  __restrict__ w1,     // [1024][4096]
             const float*  __restrict__ b1,     // [1024]
             __bf16* __restrict__ hid)          // [256][1024]
{
  __shared__ __bf16 Abuf[64 * 32];   // [m][k]
  __shared__ __bf16 Bsw[64 * 32];    // [n][k]
  const int tid    = threadIdx.x;
  const int mblock = (blockIdx.x & 3) * 64;
  const int nblock = (blockIdx.x >> 2) * 64;
  const int wave   = tid >> 5;
  const int lane   = tid & 31;
  const int mtile  = wave & 3;
  const int nhalf  = wave >> 2;
  const int lane15 = lane & 15;
  const int halfid = lane >> 4;

  const unsigned long long abase = (unsigned long long)(uintptr_t)hfeat;
  // per-thread 16B async copy slot: linear elements [tid*8, tid*8+8)
  const int a_m   = tid >> 2;
  const int a_kk0 = (tid & 3) * 8;
  const unsigned a_lds = (unsigned)(uintptr_t)(&Abuf[tid * 8]);

  v8f acc0 = {0.f, 0.f, 0.f, 0.f, 0.f, 0.f, 0.f, 0.f};
  v8f acc1 = {0.f, 0.f, 0.f, 0.f, 0.f, 0.f, 0.f, 0.f};
  for (int kd = 0; kd < 4096 / 32; ++kd) {
    __syncthreads();
    {  // A stage: async global->LDS copy, 16B per thread (bf16 passthrough)
      unsigned goff =
          (unsigned)(((size_t)(mblock + a_m) * 4096 + kd * 32 + a_kk0) * 2);
      asm volatile("global_load_async_to_lds_b128 %0, %1, %2"
                   :: "v"(a_lds), "v"(goff), "s"(abase)
                   : "memory");
    }
    {  // B stage: Bsw[nn][kk] = w1[nblock+nn][kd*32+kk]; contiguous both sides
      int nn  = tid >> 2;
      int kk0 = (tid & 3) * 8;
      const float* src = w1 + (size_t)(nblock + nn) * 4096 + kd * 32 + kk0;
#pragma unroll
      for (int i = 0; i < 8; ++i) Bsw[nn * 32 + kk0 + i] = to_bf16(src[i]);
    }
    asm volatile("s_wait_asynccnt 0x0" ::: "memory");
    __syncthreads();
    v16bf af  = gather_a(Abuf, mtile * 16 + lane15, halfid);
    v16bf bf0 = gather_b_sw(Bsw, nhalf * 32 + lane15, halfid);
    v16bf bf1 = gather_b_sw(Bsw, nhalf * 32 + 16 + lane15, halfid);
    acc0 = wmma_bf16(af, bf0, acc0);
    acc1 = wmma_bf16(af, bf1, acc1);
  }
  int   n0 = nblock + nhalf * 32 + lane15;
  int   n1 = n0 + 16;
  float bias0 = b1[n0];
  float bias1 = b1[n1];
#pragma unroll
  for (int j = 0; j < 8; ++j) {
    int m = mblock + mtile * 16 + halfid * 8 + j;
    hid[(size_t)m * D_ + n0] = to_bf16(fmaxf(acc0[j] + bias0, 0.f));
    hid[(size_t)m * D_ + n1] = to_bf16(fmaxf(acc1[j] + bias1, 0.f));
  }
}

// ---------------------------------------------------------------------------
// k4: logits = hid @ w2^T + b2   (1024 -> 3 per row).  grid = B, 128 threads
// ---------------------------------------------------------------------------
__global__ __launch_bounds__(128)
void k4_logits(const __bf16* __restrict__ hid,
               const float*  __restrict__ w2,   // [3][1024]
               const float*  __restrict__ b2,   // [3]
               float* __restrict__ logits)      // d_out
{
  __shared__ float red[NL_][128];
  int tid = threadIdx.x;
  int b   = blockIdx.x;
#pragma unroll
  for (int j = 0; j < NL_; ++j) {
    float s = 0.f;
    for (int d = tid; d < D_; d += 128)
      s += (float)hid[(size_t)b * D_ + d] * w2[j * D_ + d];
    red[j][tid] = s;
  }
  __syncthreads();
  for (int st = 64; st > 0; st >>= 1) {
    if (tid < st)
      for (int j = 0; j < NL_; ++j) red[j][tid] += red[j][tid + st];
    __syncthreads();
  }
  if (tid < NL_) logits[b * NL_ + tid] = red[tid][0] + b2[tid];
}

// ---------------------------------------------------------------------------
// launcher
// inputs: 0 c_out, 1 e_emb, 2 evidence_mask, 3 W_c, 4 W_e, 5 v,
//         6 gate_w, 7 gate_b, 8 w1, 9 b1, 10 w2, 11 b2
// d_out (float): logits [0,768) | alpha [768,66304) | head_weights [66304,67328)
// ---------------------------------------------------------------------------
extern "C" void kernel_launch(void* const* d_in, const int* in_sizes, int n_in,
                              void* d_out, int out_size, void* d_ws, size_t ws_size,
                              hipStream_t stream) {
  const float* c_out  = (const float*)d_in[0];
  const float* e_emb  = (const float*)d_in[1];
  const int*   emask  = (const int*)d_in[2];
  const float* W_c    = (const float*)d_in[3];
  const float* W_e    = (const float*)d_in[4];
  const float* v      = (const float*)d_in[5];
  const float* gate_w = (const float*)d_in[6];
  const float* gate_b = (const float*)d_in[7];
  const float* w1     = (const float*)d_in[8];
  const float* b1     = (const float*)d_in[9];
  const float* w2     = (const float*)d_in[10];
  const float* b2     = (const float*)d_in[11];

  float* out_logits = (float*)d_out;                  // B*NL
  float* out_alpha  = out_logits + B_ * NL_;          // B*H*K
  float* out_hw     = out_alpha + B_ * H_ * K_;       // B*H

  char* ws = (char*)d_ws;
  float*  wcsum = (float*)(ws + 0);                          // 4096 f32
  float*  edot  = (float*)(ws + 16384);                      // 1024 f32
  float*  eatt  = (float*)(ws + 20480);                      // B*H*D f32 (4 MB)
  __bf16* hfeat = (__bf16*)(ws + 20480 + 4194304);           // B*4D bf16 (2 MB)
  __bf16* hid   = (__bf16*)(ws + 20480 + 4194304 + 2097152); // B*D bf16 (0.5 MB)

  k0_wcsum<<<dim3((H_ * D_) / 8), dim3(256), 0, stream>>>(W_c, wcsum);
  k1_attention<<<dim3(B_ * H_), dim3(256), 0, stream>>>(
      c_out, e_emb, emask, W_e, v, gate_w, wcsum, eatt, edot, out_alpha);
  k2_gate<<<dim3(B_), dim3(256), 0, stream>>>(
      c_out, gate_w, gate_b, eatt, edot, out_hw, hfeat);
  k3_mlp1<<<dim3(64), dim3(256), 0, stream>>>(hfeat, w1, b1, hid);
  k4_logits<<<dim3(B_), dim3(128), 0, stream>>>(hid, w2, b2, out_logits);
}
